// GRU_RCN_25847113187845
// MI455X (gfx1250) — matrix operand order repaired and checked by reading, not compiled, not run on hardware
//
#include <hip/hip_runtime.h>

#define HH 56
#define WW 56
#define NPIX 3136          // 56*56
#define BB 8
#define TT 16
#define OHC 64             // hidden channels
#define PD 58              // padded spatial dim (1-px zero ring)

typedef __attribute__((ext_vector_type(16))) __bf16          v16bf;
typedef __attribute__((ext_vector_type(8)))  float           v8f;
typedef __attribute__((ext_vector_type(8)))  unsigned short  ushort8;
typedef __attribute__((ext_vector_type(4)))  unsigned int    uint4v;
typedef __attribute__((ext_vector_type(8)))  int             int8v;
typedef __attribute__((ext_vector_type(4)))  int             int4v;

union Frag { v16bf v; ushort8 u[2]; };

__device__ __forceinline__ unsigned short f2bf(float f) {
  unsigned int u = __float_as_uint(f);
  u += 0x7fffu + ((u >> 16) & 1u);      // round-to-nearest-even
  return (unsigned short)(u >> 16);
}

// ---------------------------------------------------------------------------
// TDM: 1-D tensor DMA of `nelem` bf16 elements global -> LDS.
// D# per CDNA5 ISA §8.3/8.4: group0 {count=1, lds_addr, global_addr, type=2},
// group1 {data_size=1 (2B), tensor_dim0=tile_dim0=nelem, stride0=nelem}.
// ---------------------------------------------------------------------------
__device__ __forceinline__ void tdm_load_to_lds(const void* gptr,
                                                unsigned lds_off,
                                                unsigned nelem) {
  unsigned long long ga = (unsigned long long)(uintptr_t)gptr;
  uint4v g0;
  g0[0] = 1u;                                            // count=1, user desc
  g0[1] = lds_off;                                       // lds_addr
  g0[2] = (unsigned)ga;                                  // global_addr[31:0]
  g0[3] = (unsigned)((ga >> 32) & 0x01FFFFFFu) | (2u << 30); // ga[56:32] | type=2
  int8v g1;
  g1[0] = 0x00010000;                                    // data_size=1 (2 bytes)
  g1[1] = (int)((nelem & 0xFFFFu) << 16);                // tensor_dim0[15:0]
  g1[2] = (int)((nelem >> 16) & 0xFFFFu) | (1 << 16);    // tensor_dim0[31:16], tensor_dim1=1
  g1[3] = (int)((nelem & 0xFFFFu) << 16);                // tile_dim0 = nelem
  g1[4] = 0;                                             // tile_dim1=0 (unused), tile_dim2=0
  g1[5] = (int)nelem;                                    // tensor_dim0_stride[31:0]
  g1[6] = 0;
  g1[7] = 0;
  int4v z4 = {0, 0, 0, 0};
#if __clang_major__ >= 23
  int8v z8 = {0, 0, 0, 0, 0, 0, 0, 0};
  __builtin_amdgcn_tensor_load_to_lds(g0, g1, z4, z4, z8, 0);
#else
  __builtin_amdgcn_tensor_load_to_lds(g0, g1, z4, z4, 0);
#endif
}

// ---------------------------------------------------------------------------
// Pack W [192,64,3,3], U [128,64,3,3], C [64,64,3,3] (fp32, OIHW) into bf16:
//   wpack[co][tap][ci]  co<192, ci<128 :  ci<64 -> W, ci>=64 -> U (0 for co>=128)
//   cpack[co][tap][ci]  co<64,  ci<64
// ---------------------------------------------------------------------------
__global__ void grurcn_pack_weights(const float* __restrict__ W,
                                    const float* __restrict__ U,
                                    const float* __restrict__ C,
                                    unsigned short* __restrict__ wpack,
                                    unsigned short* __restrict__ cpack)
{
  int i = blockIdx.x * blockDim.x + threadIdx.x;
  const int nW = 192 * 9 * 128;
  const int nC = 64 * 9 * 64;
  if (i < nW) {
    int ci  = i % 128;
    int t2  = i / 128;
    int tap = t2 % 9;
    int co  = t2 / 9;
    float v = 0.f;
    if (ci < 64)          v = W[(co * 64 + ci) * 9 + tap];
    else if (co < 128)    v = U[(co * 64 + (ci - 64)) * 9 + tap];
    wpack[i] = f2bf(v);
  }
  if (i < nC) {
    int ci  = i % 64;
    int t2  = i / 64;
    int tap = t2 % 9;
    int co  = t2 / 9;
    cpack[i] = f2bf(C[(co * 64 + ci) * 9 + tap]);
  }
}

// ---------------------------------------------------------------------------
// Build combined bf16 NHWC-padded input: xh[b][y+1][x+1][0:64]   = x[:,t]
//                                        xh[b][y+1][x+1][64:128] = h_prev
// Thread mapping is channel-fastest so bf16 stores are coalesced.
// ---------------------------------------------------------------------------
__global__ void grurcn_prep_xh(const float* __restrict__ x, int t,
                               const float* __restrict__ hsrc,
                               unsigned short* __restrict__ xh) // [B][58][58][128]
{
  int i = blockIdx.x * blockDim.x + threadIdx.x;
  const int N = BB * NPIX * OHC;
  if (i >= N) return;
  int ci  = i & 63;
  int pix = (i >> 6) % NPIX;
  int b   = (i >> 6) / NPIX;
  int y = pix / WW, xx = pix % WW;
  size_t base = (((size_t)b * PD + (y + 1)) * PD + (xx + 1)) * 128;
  xh[base + ci]      = f2bf(x[(((size_t)(b * TT + t) * OHC) + ci) * NPIX + pix]);
  xh[base + 64 + ci] = f2bf(hsrc[((size_t)b * OHC + ci) * NPIX + pix]);
}

// ---------------------------------------------------------------------------
// GEMM1: implicit-GEMM 3x3 conv, Cin=128 (x_t;h), Co=192 -> G fp32 (wx+uh)
// grid: (BB*HH, 3), block 128 (4 waves). Wave computes 16 co x 64 px (1 row).
// Input slab (3 padded rows, all cols/ch) is contiguous -> TDM DMA into LDS.
// ---------------------------------------------------------------------------
__global__ __launch_bounds__(128)
void grurcn_conv_gates_wmma(const unsigned short* __restrict__ xh,    // [B][58][58][128]
                            const unsigned short* __restrict__ wpack, // [192][9][128]
                            float* __restrict__ G)                    // [B][192][56][56]
{
  const int CIN = 128;
  const unsigned NEL = 3 * PD * CIN;                      // 22272 elements
  __shared__ __align__(16) unsigned short lds[(3 * PD + 8) * CIN];

  const int b       = blockIdx.x / HH;
  const int y       = blockIdx.x % HH;
  const int co_tile = blockIdx.y;               // 0..2
  const int tid     = threadIdx.x;
  const int wave    = tid >> 5;
  const int lane    = tid & 31;

  // Async DMA: each wave pulls a quarter of the slab.
  {
    const unsigned q = NEL / 4;
    const unsigned short* src = xh + ((size_t)b * PD + y) * PD * CIN + wave * q;
    unsigned lds_off = (unsigned)(uintptr_t)(void*)lds + wave * q * 2u;
    tdm_load_to_lds(src, lds_off, q);
    __builtin_amdgcn_s_wait_tensorcnt(0);
  }
  __syncthreads();

  const int co_base = co_tile * 64 + wave * 16;
  const int n       = lane & 15;
  const int kbA     = (lane >> 4) << 4;   // A frag: 0 or 16  (K 0-15 / 16-31)
  const int kbB     = (lane >> 4) << 4;   // B frag: 0 or 16  (K 0-15 / 16-31)

  v8f acc[4] = {};
  for (int tap = 0; tap < 9; ++tap) {
    const int ky = tap / 3, kx = tap % 3;
    for (int ci0 = 0; ci0 < CIN; ci0 += 32) {
      Frag A;
      const unsigned short* wp =
          wpack + ((size_t)(co_base + n) * 9 + tap) * CIN + ci0 + kbA;
      A.u[0] = *(const ushort8*)(wp);
      A.u[1] = *(const ushort8*)(wp + 8);
#pragma unroll
      for (int g = 0; g < 4; ++g) {
        const int col = g * 16 + n + kx;   // padded col index
        Frag Bf;
        const unsigned short* bp = lds + ((ky * PD + col) * CIN + ci0 + kbB);
        Bf.u[0] = *(const ushort8*)(bp);
        Bf.u[1] = *(const ushort8*)(bp + 8);
        acc[g] = __builtin_amdgcn_wmma_f32_16x16x32_bf16(
            false, A.v, false, Bf.v, (short)0, acc[g], false, false);
      }
    }
  }

  const int moff = (lane >> 4) * 8;
#pragma unroll
  for (int g = 0; g < 4; ++g) {
    int x = g * 16 + n;
    if (x < WW) {
#pragma unroll
      for (int r = 0; r < 8; ++r) {
        int co = co_base + moff + r;
        G[((size_t)(b * 192) + co) * NPIX + y * WW + x] = acc[g][r];
      }
    }
  }
}

// ---------------------------------------------------------------------------
// Gates: z = sigmoid(G_z) (stored back in place), r = sigmoid(G_r),
// rh = bf16(r * h_prev) written into padded NHWC buffer (coalesced).
// ---------------------------------------------------------------------------
__global__ void grurcn_gates(float* __restrict__ G,
                             const float* __restrict__ hprev,
                             unsigned short* __restrict__ rh)  // [B][58][58][64]
{
  int i = blockIdx.x * blockDim.x + threadIdx.x;
  const int N = BB * NPIX * OHC;
  if (i >= N) return;
  int co  = i & 63;
  int pix = (i >> 6) % NPIX;
  int b   = (i >> 6) / NPIX;
  int y = pix / WW, xx = pix % WW;
  size_t gz = ((size_t)b * 192 + co) * NPIX + pix;
  size_t gr = ((size_t)b * 192 + 64 + co) * NPIX + pix;
  float z = 1.f / (1.f + expf(-G[gz]));
  float r = 1.f / (1.f + expf(-G[gr]));
  G[gz] = z;
  float hp = hprev[((size_t)b * OHC + co) * NPIX + pix];
  rh[(((size_t)b * PD + (y + 1)) * PD + (xx + 1)) * 64 + co] = f2bf(r * hp);
}

// ---------------------------------------------------------------------------
// GEMM2: conv(rh, C) with fused GRU epilogue:
//   h_tilde = tanh(a_c + conv); h = (1-z)h + z h_tilde
// grid: (BB*HH, 1), block 128.
// ---------------------------------------------------------------------------
__global__ __launch_bounds__(128)
void grurcn_conv_cand_wmma(const unsigned short* __restrict__ rh,    // [B][58][58][64]
                           const unsigned short* __restrict__ cpack, // [64][9][64]
                           const float* __restrict__ G,              // z in [0:64], a_c in [128:192]
                           const float* __restrict__ hprev,          // [B][64][56][56]
                           float* __restrict__ out,                  // outs base [B][T][64][56][56]
                           float* __restrict__ hnext,                // [B][64][56][56]
                           int t)
{
  const int CIN = 64;
  const unsigned NEL = 3 * PD * CIN;                      // 11136 elements
  __shared__ __align__(16) unsigned short lds[(3 * PD + 8) * CIN];

  const int b    = blockIdx.x / HH;
  const int y    = blockIdx.x % HH;
  const int tid  = threadIdx.x;
  const int wave = tid >> 5;
  const int lane = tid & 31;

  {
    const unsigned q = NEL / 4;
    const unsigned short* src = rh + ((size_t)b * PD + y) * PD * CIN + wave * q;
    unsigned lds_off = (unsigned)(uintptr_t)(void*)lds + wave * q * 2u;
    tdm_load_to_lds(src, lds_off, q);
    __builtin_amdgcn_s_wait_tensorcnt(0);
  }
  __syncthreads();

  const int co_base = wave * 16;
  const int n       = lane & 15;
  const int kbA     = (lane >> 4) << 4;
  const int kbB     = (lane >> 4) << 4;

  v8f acc[4] = {};
  for (int tap = 0; tap < 9; ++tap) {
    const int ky = tap / 3, kx = tap % 3;
    for (int ci0 = 0; ci0 < CIN; ci0 += 32) {
      Frag A;
      const unsigned short* wp =
          cpack + ((size_t)(co_base + n) * 9 + tap) * CIN + ci0 + kbA;
      A.u[0] = *(const ushort8*)(wp);
      A.u[1] = *(const ushort8*)(wp + 8);
#pragma unroll
      for (int g = 0; g < 4; ++g) {
        const int col = g * 16 + n + kx;
        Frag Bf;
        const unsigned short* bp = lds + ((ky * PD + col) * CIN + ci0 + kbB);
        Bf.u[0] = *(const ushort8*)(bp);
        Bf.u[1] = *(const ushort8*)(bp + 8);
        acc[g] = __builtin_amdgcn_wmma_f32_16x16x32_bf16(
            false, A.v, false, Bf.v, (short)0, acc[g], false, false);
      }
    }
  }

  const int moff = (lane >> 4) * 8;
#pragma unroll
  for (int g = 0; g < 4; ++g) {
    int x = g * 16 + n;
    if (x < WW) {
      int pix = y * WW + x;
#pragma unroll
      for (int r = 0; r < 8; ++r) {
        int co = co_base + moff + r;
        size_t hi  = ((size_t)b * OHC + co) * NPIX + pix;
        float a_c  = G[((size_t)b * 192 + 128 + co) * NPIX + pix];
        float z    = G[((size_t)b * 192 + co) * NPIX + pix];
        float hp   = hprev[hi];
        float ht   = tanhf(a_c + acc[g][r]);
        float hn   = (1.f - z) * hp + z * ht;
        out[(((size_t)(b * TT + t)) * OHC + co) * NPIX + pix] = hn;
        hnext[hi] = hn;
      }
    }
  }
}

__global__ void grurcn_copy_hlast(const float* __restrict__ h,
                                  float* __restrict__ dst)
{
  int i = blockIdx.x * blockDim.x + threadIdx.x;
  const int N = BB * OHC * NPIX;
  if (i < N) dst[i] = h[i];
}

// ---------------------------------------------------------------------------
extern "C" void kernel_launch(void* const* d_in, const int* in_sizes, int n_in,
                              void* d_out, int out_size, void* d_ws, size_t ws_size,
                              hipStream_t stream)
{
  (void)in_sizes; (void)n_in; (void)out_size; (void)ws_size;
  const float* x  = (const float*)d_in[0];   // [8,16,64,56,56]
  const float* h0 = (const float*)d_in[1];   // [8,64,56,56]
  const float* W  = (const float*)d_in[2];   // [192,64,3,3]
  const float* U  = (const float*)d_in[3];   // [128,64,3,3]
  const float* C  = (const float*)d_in[4];   // [64,64,3,3]
  float* out = (float*)d_out;                // outs [8,16,64,56,56] ++ h_last [8,64,56,56]

  char* ws = (char*)d_ws;
  size_t off = 0;
  auto alloc = [&](size_t bytes) -> void* {
    off = (off + 255) & ~(size_t)255;
    void* p = ws + off;
    off += bytes;
    return p;
  };
  const size_t xhBytes = (size_t)BB * PD * PD * 128 * 2;   // 6.9 MB (padded NHWC)
  const size_t rhBytes = (size_t)BB * PD * PD * 64 * 2;    // 3.4 MB (padded NHWC)
  unsigned short* xh    = (unsigned short*)alloc(xhBytes);
  unsigned short* rh    = (unsigned short*)alloc(rhBytes);
  float*          G     = (float*)         alloc((size_t)BB * 192 * NPIX * 4); // 19.3 MB
  float*          hbuf  = (float*)         alloc((size_t)BB * OHC * NPIX * 4); // 6.4 MB
  unsigned short* wpack = (unsigned short*)alloc((size_t)192 * 9 * 128 * 2);
  unsigned short* cpack = (unsigned short*)alloc((size_t)64 * 9 * 64 * 2);

  // Zero halo rings (and interiors) once per launch; prep/gates rewrite interiors.
  hipMemsetAsync(xh, 0, xhBytes, stream);
  hipMemsetAsync(rh, 0, rhBytes, stream);

  {
    const int nW = 192 * 9 * 128;
    grurcn_pack_weights<<<(nW + 255) / 256, 256, 0, stream>>>(W, U, C, wpack, cpack);
  }

  const int Nel = BB * OHC * NPIX;
  dim3 gelem((Nel + 255) / 256);
  dim3 ggemm1(BB * HH, 3);
  dim3 ggemm2(BB * HH, 1);

  for (int t = 0; t < TT; ++t) {
    const float* hprev = (t == 0) ? h0 : hbuf;
    grurcn_prep_xh<<<gelem, 256, 0, stream>>>(x, t, hprev, xh);
    grurcn_conv_gates_wmma<<<ggemm1, 128, 0, stream>>>(xh, wpack, G);
    grurcn_gates<<<gelem, 256, 0, stream>>>(G, hprev, rh);
    grurcn_conv_cand_wmma<<<ggemm2, 128, 0, stream>>>(rh, cpack, G, hprev, out, hbuf, t);
  }
  grurcn_copy_hlast<<<gelem, 256, 0, stream>>>(hbuf, out + (size_t)BB * TT * OHC * NPIX);
}
